// ATKT_3487513444378
// MI455X (gfx1250) — matrix operand order repaired and checked by reading, not compiled, not run on hardware
//
#include <hip/hip_runtime.h>

// ---------------------------------------------------------------------------
// ATKT forward for MI455X (gfx1250, wave32, WMMA).
//   d_out = [ res (64*512*1000 f32) | sa_emb (64*512*512 f32) ]
// v4: double-buffered async LDS staging in the GEMM (DMA overlaps WMMA),
//     A-tiles preloaded before the WMMA burst, LSTM xproj add deferred to
//     epilogue so global loads hide under the recurrent WMMAs.
// ---------------------------------------------------------------------------

typedef __attribute__((ext_vector_type(16))) _Float16 v16h;
typedef __attribute__((ext_vector_type(8)))  float    v8f;
typedef __attribute__((ext_vector_type(4)))  int      vi4;

#define Bsz   64
#define Seq   512
#define HID   256
#define LIN   512        // LSTM input dim
#define G4    1024       // 4*HID
#define ADIM  80
#define NC    1000
#define NCPAD 1024       // fc_W rows padded to 16 macro-N tiles of 64
#define MLPAD 128        // mlp_W rows padded to 2 macro-N tiles of 64
#define ROWS  (Bsz*Seq)  // 32768

#define LDSPAD 40        // GEMM LDS row stride (halves): 20 dwords, conflict-free
#define HPAD   264       // LSTM LDS row stride (halves): 132 dwords, conflict-free

// ---------------- async global->LDS (gfx1250) with sync fallback ------------
#define ATKT_TRY_ASYNC 1
#if ATKT_TRY_ASYNC && __has_builtin(__builtin_amdgcn_global_load_async_to_lds_b128) && \
    __has_builtin(__builtin_amdgcn_s_wait_asynccnt)
#define ATKT_ASYNC 1
#else
#define ATKT_ASYNC 0
#endif

#define AS_GLOBAL __attribute__((address_space(1)))
#define AS_LDS    __attribute__((address_space(3)))

__device__ __forceinline__ void copy16B_to_lds(_Float16* l, const _Float16* g) {
#if ATKT_ASYNC
  __builtin_amdgcn_global_load_async_to_lds_b128(
      (AS_GLOBAL vi4*)g, (AS_LDS vi4*)l, 0, 0);
#else
  *(uint4*)l = *(const uint4*)g;
#endif
}
__device__ __forceinline__ void lds_copy_join() {
#if ATKT_ASYNC
  __builtin_amdgcn_s_wait_asynccnt(0);
#endif
}

// ---------------- WMMA tile loaders (f16 16x16x32, wave32 layouts) ----------
// A 16x32: lane L holds row M=L%16; lanes<16 carry K 0..7,16..23; lanes>=16 K 8..15,24..31
// B 32x16: lane L holds col N=L%16; lanes<16 carry K 0..7,16..23; lanes>=16 K 8..15,24..31
// C 16x16 f32: VGPR r: lanes<16 -> M=r, lanes>=16 -> M=r+8; N=lane%16
__device__ __forceinline__ v16h load_a_tile(const _Float16* __restrict__ arow, int kgrp) {
  v16h a;
  const _Float16* ap = arow + kgrp * 8;
#pragma unroll
  for (int e = 0; e < 8; ++e) a[e] = ap[e];
#pragma unroll
  for (int e = 0; e < 8; ++e) a[8 + e] = ap[16 + e];
  return a;
}
__device__ __forceinline__ v16h load_b_tile(const _Float16* __restrict__ wrow, int kgrp) {
  v16h b;
  const _Float16* bp = wrow + kgrp * 8;
#pragma unroll
  for (int e = 0; e < 8; ++e) b[e] = bp[e];
#pragma unroll
  for (int e = 0; e < 8; ++e) b[8 + e] = bp[16 + e];
  return b;
}

// ---------------- weight conversion kernels --------------------------------
__global__ void cvt_f16_kernel(const float* __restrict__ src, _Float16* __restrict__ dst, int n) {
  int i = blockIdx.x * blockDim.x + threadIdx.x;
  if (i < n) dst[i] = (_Float16)src[i];
}
__global__ void cvt_pad_kernel(const float* __restrict__ src, _Float16* __restrict__ dst,
                               int srows, int drows, int cols) {
  int i = blockIdx.x * blockDim.x + threadIdx.x;
  if (i >= drows * cols) return;
  int r = i / cols, c = i % cols;
  dst[i] = (r < srows) ? (_Float16)src[r * cols + c] : (_Float16)0.0f;
}
__global__ void bias_sum_kernel(const float* __restrict__ a, const float* __restrict__ b,
                                float* __restrict__ o, int n) {
  int i = blockIdx.x * blockDim.x + threadIdx.x;
  if (i < n) o[i] = a[i] + b[i];
}

// ---------------- embedding gather -----------------------------------------
__global__ __launch_bounds__(256) void embed_kernel(
    const int* __restrict__ skill, const int* __restrict__ answer,
    const float* __restrict__ s_emb, const float* __restrict__ a_emb,
    float* __restrict__ sa_out,          // [b][s][512] f32 (2nd output)
    _Float16* __restrict__ x16)          // [s][b][512] f16 (GEMM/LSTM layout)
{
  int bs = blockIdx.x;                   // 0..32767
  int b = bs >> 9, s = bs & 511;
  int an = answer[bs];
  const float* se = s_emb + (size_t)skill[bs] * 256;
  const float* ae = a_emb + (size_t)an * 256;
  bool sa_first = (an == 1);
  for (int c = threadIdx.x; c < 512; c += blockDim.x) {
    float v;
    if (sa_first) v = (c < 256) ? se[c] : ae[c - 256];
    else          v = (c < 256) ? ae[c] : se[c - 256];
    sa_out[(size_t)bs * 512 + c] = v;
    x16[((size_t)s * Bsz + b) * 512 + c] = (_Float16)v;
  }
}

// ---------------- 64x64 macro-tile WMMA GEMM:  C = act(A @ W^T + bias) ------
// A: [M][K] f16 (lda), W: [N][K] f16 (ldw). Block = 128 threads (4 waves).
// Wave w owns N-tile w; each wave computes 4 M-tiles (4x v8f acc).
// Double-buffered LDS K-strips: async DMA of strip k+32 overlaps WMMA on k.
__global__ __launch_bounds__(128) void wmma_gemm64_kernel(
    const _Float16* __restrict__ A, int lda,
    const _Float16* __restrict__ W, int ldw,
    const float* __restrict__ bias,
    float* __restrict__ Cf32, _Float16* __restrict__ Cf16,
    int ldc, int Nlim, int K, int act)
{
  __shared__ __align__(16) _Float16 Ash[2][64 * LDSPAD];
  __shared__ __align__(16) _Float16 Bsh[2][64 * LDSPAD];
  const int nt0 = blockIdx.x * 64, mt0 = blockIdx.y * 64;
  const int tid = threadIdx.x;
  const int wave = tid >> 5, lane = tid & 31;
  const int ml = lane & 15, kg = lane >> 4;

  // per-thread staging coordinates: 64 rows x 4 chunks of 8 halves, x2 strips
  const int r0 = tid >> 1, c0 = (tid & 1) * 16;            // chunks 0..1 of row r0
  const _Float16* Arow0 = &A[((size_t)mt0 + r0) * lda + c0];
  const _Float16* Wrow0 = &W[((size_t)nt0 + r0) * ldw + c0];
  _Float16* AshT = &Ash[0][r0 * LDSPAD + c0];
  _Float16* BshT = &Bsh[0][r0 * LDSPAD + c0];

#define STAGE(buf, kk)                                              \
  do {                                                              \
    copy16B_to_lds(AshT + (buf) * 64 * LDSPAD,     Arow0 + (kk));   \
    copy16B_to_lds(AshT + (buf) * 64 * LDSPAD + 8, Arow0 + (kk) + 8);\
    copy16B_to_lds(BshT + (buf) * 64 * LDSPAD,     Wrow0 + (kk));   \
    copy16B_to_lds(BshT + (buf) * 64 * LDSPAD + 8, Wrow0 + (kk) + 8);\
  } while (0)

  v8f acc[4] = {{}, {}, {}, {}};
  int cur = 0;
  STAGE(0, 0);
  lds_copy_join();
  __syncthreads();
  for (int k = 0; k < K; k += 32) {
    if (k + 32 < K) STAGE(cur ^ 1, k + 32);   // async prefetch next strip
    // preload all operand tiles, then burst the 4 WMMAs
    v16h b  = load_b_tile(&Bsh[cur][(wave * 16 + ml) * LDSPAD], kg);
    v16h a0 = load_a_tile(&Ash[cur][(0 * 16 + ml) * LDSPAD], kg);
    v16h a1 = load_a_tile(&Ash[cur][(1 * 16 + ml) * LDSPAD], kg);
    v16h a2 = load_a_tile(&Ash[cur][(2 * 16 + ml) * LDSPAD], kg);
    v16h a3 = load_a_tile(&Ash[cur][(3 * 16 + ml) * LDSPAD], kg);
    acc[0] = __builtin_amdgcn_wmma_f32_16x16x32_f16(false, a0, false, b, (short)0, acc[0], false, false);
    acc[1] = __builtin_amdgcn_wmma_f32_16x16x32_f16(false, a1, false, b, (short)0, acc[1], false, false);
    acc[2] = __builtin_amdgcn_wmma_f32_16x16x32_f16(false, a2, false, b, (short)0, acc[2], false, false);
    acc[3] = __builtin_amdgcn_wmma_f32_16x16x32_f16(false, a3, false, b, (short)0, acc[3], false, false);
    lds_copy_join();       // next strip landed
    __syncthreads();       // everyone done reading cur / sees next
    cur ^= 1;
  }
#undef STAGE

  const int n = nt0 + wave * 16 + ml;
  const float bv = (bias && n < Nlim) ? bias[n] : 0.0f;
#pragma unroll
  for (int mt = 0; mt < 4; ++mt) {
#pragma unroll
    for (int r = 0; r < 8; ++r) {
      float v = acc[mt][r] + bv;
      if (act == 1)      v = 1.0f / (1.0f + __expf(-v));   // sigmoid
      else if (act == 2) v = tanhf(v);                     // tanh
      const size_t row = (size_t)mt0 + mt * 16 + kg * 8 + r;
      if (n < Nlim) {
        if (Cf32) Cf32[row * (size_t)ldc + n] = v;
        if (Cf16) Cf16[row * (size_t)ldc + n] = (_Float16)v;
      }
    }
  }
}

// ---------------- grid barrier (generation counter + s_sleep) ---------------
__device__ __forceinline__ void grid_sync(unsigned* bar, unsigned nwg) {
  __syncthreads();
  if (threadIdx.x == 0) {
    __threadfence();
    unsigned gen  = __hip_atomic_load(bar + 1, __ATOMIC_ACQUIRE, __HIP_MEMORY_SCOPE_AGENT);
    unsigned prev = __hip_atomic_fetch_add(bar, 1u, __ATOMIC_ACQ_REL, __HIP_MEMORY_SCOPE_AGENT);
    if (prev == nwg - 1u) {
      __hip_atomic_store(bar, 0u, __ATOMIC_RELAXED, __HIP_MEMORY_SCOPE_AGENT);
      __hip_atomic_fetch_add(bar + 1, 1u, __ATOMIC_RELEASE, __HIP_MEMORY_SCOPE_AGENT);
    } else {
      while (__hip_atomic_load(bar + 1, __ATOMIC_ACQUIRE, __HIP_MEMORY_SCOPE_AGENT) == gen)
        __builtin_amdgcn_s_sleep(2);
    }
  }
  __syncthreads();
}

// ---------------- persistent LSTM recurrence -------------------------------
// 16 WGs x 512 threads (16 waves). WG `wg` owns hidden cols [wg*16, wg*16+16),
// computing all 4 gate tiles for them. Its 64x256 Whh slice stays in LDS.
// Dynamic LDS: hsh(64xHPAD f16) | gsh(64x64 f32) | wsh(64xHPAD f16) = 84 KB.
#define NWG 16
__global__ __launch_bounds__(512) void lstm_kernel(
    const _Float16* __restrict__ xproj,  // [512][64][1024] f16
    const _Float16* __restrict__ Whh16,  // [1024][256] f16
    float* __restrict__ cbuf,            // [64][256]
    _Float16* __restrict__ hbuf,         // [64][256] f16
    float* __restrict__ out32,           // [b*512+t][256]
    _Float16* __restrict__ out16,        // same layout, f16
    unsigned* __restrict__ bar)
{
  extern __shared__ __align__(16) char smem[];
  _Float16* hsh = (_Float16*)smem;                            // 64*HPAD*2 = 33792 B
  float*    gsh = (float*)(smem + 64 * HPAD * 2);             // 64*64*4   = 16384 B
  _Float16* wsh = (_Float16*)(smem + 64 * HPAD * 2 + 64 * 64 * 4);  // 33792 B

  const int tid = threadIdx.x;
  const int wg  = blockIdx.x;

  // zero my slice of c/h state
  for (int p = tid; p < Bsz * 16; p += 512) {
    int b = p >> 4, cl = p & 15, hc = wg * 16 + cl;
    cbuf[b * HID + hc] = 0.0f;
    hbuf[b * HID + hc] = (_Float16)0.0f;
  }
  // cache this WG's Whh slice: chunkRow r -> global row (r>>4)*256 + wg*16 + (r&15)
#pragma unroll
  for (int i = 0; i < 4; ++i) {
    int cid = tid + i * 512;           // 0..2047
    int r = cid >> 5, c8 = (cid & 31) * 8;
    int grow = (r >> 4) * HID + wg * 16 + (r & 15);
    copy16B_to_lds(&wsh[r * HPAD + c8], &Whh16[(size_t)grow * HID + c8]);
  }
  lds_copy_join();
  grid_sync(bar, NWG);

  const int lane = tid & 31;
  const int wave = tid >> 5;     // 0..15
  const int mt   = wave & 3;     // batch tile (16 rows)
  const int gt   = wave >> 2;    // gate 0..3
  const int ml   = lane & 15;
  const int kg   = lane >> 4;
  const int ncol = gt * HID + wg * 16;               // global gate-column base
  const _Float16* wrow  = wsh + (gt * 16 + ml) * HPAD;
  const _Float16* hrow0 = hsh + (mt * 16 + ml) * HPAD;

  for (int t = 0; t < Seq; ++t) {
    // stage full h (64x256 f16) into LDS, padded rows
#pragma unroll
    for (int i = 0; i < 4; ++i) {
      int cid = tid + i * 512;         // 0..2047
      int r = cid >> 5, c8 = (cid & 31) * 8;
      copy16B_to_lds(&hsh[r * HPAD + c8], &hbuf[r * HID + c8]);
    }
    // xproj tile loads issue now; latency hides under the WMMA burst below
    float xv[8];
    {
      const _Float16* xp = xproj + ((size_t)t * Bsz + mt * 16) * G4 + ncol;
#pragma unroll
      for (int r = 0; r < 8; ++r)
        xv[r] = (float)xp[(size_t)(kg * 8 + r) * G4 + ml];
    }
    lds_copy_join();
    __syncthreads();

    // gates = h @ Whh^T  (K = 256, 8 WMMAs; A from hsh, B from wsh)
    v8f acc = {};
#pragma unroll
    for (int k = 0; k < HID; k += 32) {
      v16h a = load_a_tile(hrow0 + k, kg);
      v16h b = load_b_tile(wrow + k, kg);
      acc = __builtin_amdgcn_wmma_f32_16x16x32_f16(false, a, false, b, (short)0, acc,
                                                   false, false);
    }
#pragma unroll
    for (int r = 0; r < 8; ++r)
      gsh[(mt * 16 + kg * 8 + r) * 64 + gt * 16 + ml] = acc[r] + xv[r];
    __syncthreads();

    // elementwise cell update for our 64x16 slice (2 per thread)
#pragma unroll
    for (int i = 0; i < 2; ++i) {
      int p = tid + i * 512;
      int b = p >> 4, cl = p & 15, hc = wg * 16 + cl;
      float iv = gsh[b * 64 + cl];
      float fv = gsh[b * 64 + 16 + cl];
      float gv = gsh[b * 64 + 32 + cl];
      float ov = gsh[b * 64 + 48 + cl];
      iv = 1.0f / (1.0f + __expf(-iv));
      fv = 1.0f / (1.0f + __expf(-fv));
      ov = 1.0f / (1.0f + __expf(-ov));
      gv = tanhf(gv);
      float c = fv * cbuf[b * HID + hc] + iv * gv;
      cbuf[b * HID + hc] = c;
      float h = ov * tanhf(c);
      hbuf[b * HID + hc] = (_Float16)h;
      size_t row = (size_t)b * Seq + t;
      out32[row * HID + hc] = h;
      out16[row * HID + hc] = (_Float16)h;
    }
    grid_sync(bar, NWG);   // publish h to all WGs before next step
  }
}

// ---------------- attention logits -> per-batch exp + prefix-Z -------------
__global__ __launch_bounds__(512) void att_kernel(
    const float* __restrict__ tanhbuf,  // [32768][80]
    const float* __restrict__ simW,     // [80]
    float* __restrict__ ebuf, float* __restrict__ zbuf)  // [64][512] each
{
  __shared__ float red[512];
  __shared__ float sh[512];
  int b = blockIdx.x, s = threadIdx.x;
  const float* r = tanhbuf + ((size_t)b * Seq + s) * ADIM;
  float dot = 0.0f;
#pragma unroll 8
  for (int i = 0; i < ADIM; ++i) dot += r[i] * simW[i];
  red[s] = dot; __syncthreads();
  for (int off = 256; off > 0; off >>= 1) {
    if (s < off) red[s] = fmaxf(red[s], red[s + off]);
    __syncthreads();
  }
  float m = red[0];
  __syncthreads();
  float e = __expf(dot - m);
  sh[s] = e; __syncthreads();
  for (int off = 1; off < 512; off <<= 1) {  // Hillis-Steele inclusive scan
    float v = (s >= off) ? sh[s - off] : 0.0f;
    __syncthreads();
    sh[s] += v;
    __syncthreads();
  }
  ebuf[(size_t)b * Seq + s] = e;
  zbuf[(size_t)b * Seq + s] = sh[s];
}

// ---------------- attention as prefix scan + build `final` -----------------
// attn_out[s] = prefix(e*out)/Z_s ; attn_cum_1[s] = sum_{k<s} attn_out[k]
__global__ __launch_bounds__(256) void final_scan_kernel(
    const float* __restrict__ ebuf, const float* __restrict__ zbuf,
    const float* __restrict__ out32,
    _Float16* __restrict__ final16)     // [32768][512] = [cum | out]
{
  int b = blockIdx.x, h = threadIdx.x;
  float acc = 0.0f, cum = 0.0f;
  for (int s = 0; s < Seq; ++s) {
    size_t row = (size_t)b * Seq + s;
    float ev = ebuf[(size_t)b * Seq + s];
    float zv = zbuf[(size_t)b * Seq + s];
    float ov = out32[row * HID + h];
    final16[row * 512 + h]       = (_Float16)cum;  // exclusive cumsum of attn_out
    final16[row * 512 + HID + h] = (_Float16)ov;
    acc += ev * ov;
    cum += acc / zv;
  }
}

// ---------------------------------------------------------------------------
static inline char* bump(char*& p, size_t bytes) {
  char* r = p;
  p += (bytes + 255) & ~(size_t)255;
  return r;
}

extern "C" void kernel_launch(void* const* d_in, const int* in_sizes, int n_in,
                              void* d_out, int out_size, void* d_ws, size_t ws_size,
                              hipStream_t stream) {
  const int*   skill   = (const int*)d_in[0];
  const int*   answer  = (const int*)d_in[1];
  const float* s_emb   = (const float*)d_in[2];
  const float* a_emb   = (const float*)d_in[3];
  const float* Wih     = (const float*)d_in[4];
  const float* Whh     = (const float*)d_in[5];
  const float* bih     = (const float*)d_in[6];
  const float* bhh     = (const float*)d_in[7];
  const float* mlpW    = (const float*)d_in[8];
  const float* mlpb    = (const float*)d_in[9];
  const float* simW    = (const float*)d_in[10];
  const float* fcW     = (const float*)d_in[11];
  const float* fcb     = (const float*)d_in[12];
  (void)in_sizes; (void)n_in; (void)out_size; (void)ws_size;

  float* res    = (float*)d_out;                       // [32768][1000]
  float* sa_out = res + (size_t)ROWS * NC;             // [32768][512]

  // workspace carve-up (~198 MB)
  char* p = (char*)d_ws;
  _Float16* x16     = (_Float16*)bump(p, (size_t)ROWS * LIN * 2);
  _Float16* Wih16   = (_Float16*)bump(p, (size_t)G4 * LIN * 2);
  _Float16* Whh16   = (_Float16*)bump(p, (size_t)G4 * HID * 2);
  _Float16* mlp16   = (_Float16*)bump(p, (size_t)MLPAD * HID * 2);
  _Float16* fc16    = (_Float16*)bump(p, (size_t)NCPAD * LIN * 2);
  float*    bsum    = (float*)   bump(p, (size_t)G4 * 4);
  _Float16* xproj16 = (_Float16*)bump(p, (size_t)Seq * Bsz * G4 * 2);
  float*    cbuf    = (float*)   bump(p, (size_t)Bsz * HID * 4);
  _Float16* hbuf    = (_Float16*)bump(p, (size_t)Bsz * HID * 2);
  float*    out32   = (float*)   bump(p, (size_t)ROWS * HID * 4);
  _Float16* out16   = (_Float16*)bump(p, (size_t)ROWS * HID * 2);
  float*    tanhbuf = (float*)   bump(p, (size_t)ROWS * ADIM * 4);
  float*    ebuf    = (float*)   bump(p, (size_t)Bsz * Seq * 4);
  float*    zbuf    = (float*)   bump(p, (size_t)Bsz * Seq * 4);
  _Float16* final16 = (_Float16*)bump(p, (size_t)ROWS * 512 * 2);
  unsigned* bar     = (unsigned*)bump(p, 256);

  // K1: weight conversion (pad mlp_W 80->128 rows, fc_W 1000->1024 rows)
  cvt_f16_kernel<<<(G4 * LIN + 255) / 256, 256, 0, stream>>>(Wih, Wih16, G4 * LIN);
  cvt_f16_kernel<<<(G4 * HID + 255) / 256, 256, 0, stream>>>(Whh, Whh16, G4 * HID);
  cvt_pad_kernel<<<(MLPAD * HID + 255) / 256, 256, 0, stream>>>(mlpW, mlp16, ADIM, MLPAD, HID);
  cvt_pad_kernel<<<(NCPAD * LIN + 255) / 256, 256, 0, stream>>>(fcW, fc16, NC, NCPAD, LIN);
  bias_sum_kernel<<<(G4 + 255) / 256, 256, 0, stream>>>(bih, bhh, bsum, G4);

  // K2: embedding gather (-> sa_emb output + x_f16 [s][b][512])
  embed_kernel<<<ROWS, 256, 0, stream>>>(skill, answer, s_emb, a_emb, sa_out, x16);

  // K3: xproj = x @ Wih^T + (bih+bhh), rows ordered [s][b]
  wmma_gemm64_kernel<<<dim3(G4 / 64, ROWS / 64), 128, 0, stream>>>(
      x16, LIN, Wih16, LIN, bsum, nullptr, xproj16, G4, G4, LIN, 0);

  // K4: persistent LSTM (grid barrier counters must start at 0)
  (void)hipMemsetAsync(bar, 0, 2 * sizeof(unsigned), stream);
  size_t lstm_lds = (size_t)64 * HPAD * 2 + 64 * 64 * 4 + (size_t)64 * HPAD * 2; // 83968 B
  lstm_kernel<<<NWG, 512, lstm_lds, stream>>>(xproj16, Whh16, cbuf, hbuf, out32, out16, bar);

  // K5: tanh(out @ mlp_W^T + mlp_b), N padded to 128, store 80
  wmma_gemm64_kernel<<<dim3(MLPAD / 64, ROWS / 64), 128, 0, stream>>>(
      out16, HID, mlp16, HID, mlpb, tanhbuf, nullptr, ADIM, ADIM, HID, 2);

  // K6: att logits + per-batch softmax prefix terms
  att_kernel<<<Bsz, 512, 0, stream>>>(tanhbuf, simW, ebuf, zbuf);

  // K7: attention prefix scan -> final_f16 = [attn_cum_1 | out]
  final_scan_kernel<<<Bsz, 256, 0, stream>>>(ebuf, zbuf, out32, final16);

  // K8: res = sigmoid(final @ fc_W^T + fc_b), N padded to 1024, store 1000
  wmma_gemm64_kernel<<<dim3(NCPAD / 64, ROWS / 64), 128, 0, stream>>>(
      final16, 512, fc16, LIN, fcb, res, nullptr, NC, NC, LIN, 1);
}
